// QFeedForward_65481071403104
// MI455X (gfx1250) — matrix operand order repaired and checked by reading, not compiled, not run on hardware
//
#include <hip/hip_runtime.h>

typedef __attribute__((ext_vector_type(16))) _Float16 v16h;
typedef __attribute__((ext_vector_type(8)))  _Float16 v8h;
typedef __attribute__((ext_vector_type(8)))  float    v8f;
typedef __attribute__((ext_vector_type(4)))  float    v4f;
typedef __attribute__((ext_vector_type(4)))  unsigned int u32x4;
typedef __attribute__((ext_vector_type(8)))  int      i32x8;
typedef __attribute__((ext_vector_type(4)))  int      i32x4;

#define TOKENS  32768   // B*S
#define E_DIM   512
#define FFN_DIM 2048

#define BM 128          // rows per workgroup (8 waves x 16)
#define BN 128          // cols per workgroup
#define BK 32           // K chunk = wmma f16 K
#define BSTRIDE (BK + 8)   // halves: 80B row stride (64B data + 16B TDM pad)

#if defined(__has_builtin)
# if __has_builtin(__builtin_amdgcn_tensor_load_to_lds) && \
     __has_builtin(__builtin_amdgcn_s_wait_tensorcnt)
#  define HAVE_TDM 1
# endif
#endif
#ifndef HAVE_TDM
# define HAVE_TDM 0
#endif

// B fragment: lane col N = l16; element j -> K = 16*hk + j (contiguous in LDS)
__device__ __attribute__((always_inline))
static v16h load_bfrag(const _Float16* bbuf, int c, int l16, int hk)
{
    const _Float16* bp = bbuf + (c * 16 + l16) * BSTRIDE + 16 * hk;
    v8h lo = *(const v8h*)bp;                // ds_load_b128
    v8h hi = *(const v8h*)(bp + 8);          // ds_load_b128
    return __builtin_shufflevector(lo, hi,
           0,1,2,3,4,5,6,7,8,9,10,11,12,13,14,15);
}

// ---- one K-step of the fused GEMM: A generated in registers in WMMA layout,
// B fragments pipelined 3-deep from (padded) LDS, 8 WMMAs -> 16x128 of C. ----
__device__ __attribute__((always_inline))
static void mma_step(const float* sW1, const _Float16* bbuf,
                     const float meas[4], int kc, int hk, int l16, v8f acc[8])
{
    // A fragment: lane row M = l16; element j -> K = 16*hk + j
    v16h afr;
#pragma unroll
    for (int j = 0; j < 16; ++j) {
        const int k = kc + 16 * hk + j;
        v4f w1r = ((const v4f*)sW1)[k];          // ds_load_b128, broadcast
        float h = meas[0] * w1r[0];
        h = fmaf(meas[1], w1r[1], h);
        h = fmaf(meas[2], w1r[2], h);
        h = fmaf(meas[3], w1r[3], h);
        afr[j] = (_Float16)fmaxf(h, 0.0f);       // relu + cvt
    }

    // 3-deep register pipeline over the 8 B fragments: loads for fragment c+2
    // are in flight while wmma(c) executes -> dscnt waits stay non-zero.
    v16h b0 = load_bfrag(bbuf, 0, l16, hk);
    v16h b1 = load_bfrag(bbuf, 1, l16, hk);
#pragma unroll
    for (int c = 0; c < 8; ++c) {
        v16h bn = b1;
        if (c + 2 < 8) bn = load_bfrag(bbuf, c + 2, l16, hk);
        acc[c] = __builtin_amdgcn_wmma_f32_16x16x32_f16(
            false, afr, false, b0, (short)0, acc[c], false, false);
        b0 = b1;
        b1 = bn;
    }
}

#if HAVE_TDM
// Issue a TDM 2D tile load: 32(K,f16) x 128(E) from W2h into LDS, with TDM
// padding 16B after each 64B row -> 80B LDS row stride (= BSTRIDE halves).
__device__ __attribute__((always_inline))
static void tdm_issue_b(unsigned long long gaddr, unsigned lds_addr)
{
    u32x4 g0 = {0u, 0u, 0u, 0u};
    g0[0] = 1u;                                   // count=1 (valid descriptor)
    g0[1] = lds_addr;                             // LDS byte address
    g0[2] = (unsigned)(gaddr & 0xffffffffull);    // global_addr[31:0]
    g0[3] = (unsigned)((gaddr >> 32) & 0x1ffffffull) | (2u << 30); // [56:32]+type=2

    i32x8 g1 = {0, 0, 0, 0, 0, 0, 0, 0};
    g1[0] = (1 << 16)      // data_size = 2B (f16)
          | (1 << 20)      // pad_enable
          | (3 << 22)      // pad_interval: 16 DWORDs (one 64B tile row)
          | (3 << 25);     // pad_amount:   4 DWORDs (16B)
    g1[1] = (int)(FFN_DIM << 16);   // tensor_dim0 = 2048 (low16 at bits 63:48)
    g1[2] = (int)(E_DIM << 16);     // tensor_dim1 = 512
    g1[3] = (int)(BK << 16);        // tile_dim0 = 32 elements
    g1[4] = BN;                     // tile_dim1 = 128 rows
    g1[5] = FFN_DIM;                // tensor_dim0_stride = 2048 elements
    i32x4 g2 = {0, 0, 0, 0};
    i32x4 g3 = {0, 0, 0, 0};
#if __clang_major__ >= 23
    i32x8 g4 = {0, 0, 0, 0, 0, 0, 0, 0};
    __builtin_amdgcn_tensor_load_to_lds(g0, g1, g2, g3, g4, 0);
#else
    __builtin_amdgcn_tensor_load_to_lds(g0, g1, g2, g3, 0);
#endif
}
#endif

// ---------------- main fused kernel (f16 W2 source in d_ws) ----------------
__global__ __launch_bounds__(256)
void qffn_wmma_tdm(const float* __restrict__ x, const float* __restrict__ theta,
                   const float* __restrict__ W1, const _Float16* __restrict__ W2h,
                   float* __restrict__ out)
{
    __shared__ float    sW1[FFN_DIM * 4];          // 32 KB
    __shared__ _Float16 sB[2][BN * BSTRIDE];       // 2 x 10 KB, TDM-filled

    const int tid  = threadIdx.x;
    const int lane = tid & 31;
    const int l16  = lane & 15;
    const int hk   = lane >> 4;
    const int wave = tid >> 5;                     // 0..7 -> 16-row slab

    const int m0 = blockIdx.x * BM;
    const int n0 = blockIdx.y * BN;

    for (int i = tid; i < FFN_DIM; i += 256)       // stage W1 f32 in LDS
        ((v4f*)sW1)[i] = ((const v4f*)W1)[i];

    // measurement: <Z_w> = cos(theta_w) * cos(x[n,w])
    float meas[4];
    {
        const int m = m0 + wave * 16 + l16;
        v4f xr = *(const v4f*)(x + (size_t)m * E_DIM);
        meas[0] = __cosf(xr[0]) * __cosf(theta[0]);
        meas[1] = __cosf(xr[1]) * __cosf(theta[1]);
        meas[2] = __cosf(xr[2]) * __cosf(theta[2]);
        meas[3] = __cosf(xr[3]) * __cosf(theta[3]);
    }

    v8f acc[8];
#pragma unroll
    for (int c = 0; c < 8; ++c)
#pragma unroll
        for (int r = 0; r < 8; ++r) acc[c][r] = 0.0f;

#if HAVE_TDM
    const unsigned long long gbase =
        (unsigned long long)(uintptr_t)(W2h + (size_t)n0 * FFN_DIM);
    const unsigned lds0 = (unsigned)(uintptr_t)&sB[0][0];
    const unsigned lds1 = (unsigned)(uintptr_t)&sB[1][0];

    if (wave == 0) tdm_issue_b(gbase, lds0);       // prime buffer 0
    int i = 0;
    for (int kc = 0; kc < FFN_DIM; kc += BK, ++i) {
        if (wave == 0) {
            if (kc + BK < FFN_DIM) {               // launch next chunk, then
                tdm_issue_b(gbase + (unsigned long long)(kc + BK) * 2,
                            (i & 1) ? lds0 : lds1);
                __builtin_amdgcn_s_wait_tensorcnt(1);  // current chunk done
            } else {
                __builtin_amdgcn_s_wait_tensorcnt(0);
            }
        }
        __syncthreads();                           // publish buf[i&1]
        mma_step(sW1, sB[i & 1], meas, kc, hk, l16, acc);
        __syncthreads();                           // reads done before reuse
    }
#else
    // manual f16 staging fallback (no TDM builtin on this toolchain)
    const _Float16* wsrc = W2h + (size_t)(n0 + (tid >> 1)) * FFN_DIM + (tid & 1) * 16;
    _Float16* bdst = &sB[0][(tid >> 1) * BSTRIDE + (tid & 1) * 16];
    for (int kc = 0; kc < FFN_DIM; kc += BK) {
        __syncthreads();
        v8h lo = *(const v8h*)(wsrc + kc);
        v8h hi = *(const v8h*)(wsrc + kc + 8);
        *(v8h*)bdst = lo;
        *(v8h*)(bdst + 8) = hi;
        __syncthreads();
        mma_step(sW1, sB[0], meas, kc, hk, l16, acc);
    }
#endif

    // store C: element (r, lane) -> (M = r + 8*hk, N = l16); NT: write-once out
    const int mb = m0 + wave * 16 + 8 * hk;
#pragma unroll
    for (int c = 0; c < 8; ++c) {
        const int n = n0 + c * 16 + l16;
#pragma unroll
        for (int r = 0; r < 8; ++r)
            __builtin_nontemporal_store(acc[c][r],
                &out[(size_t)(mb + r) * E_DIM + n]);
    }
}

// -------- fallback if workspace too small: stage f32 W2, convert in-loop ----
__global__ __launch_bounds__(256)
void qffn_wmma_f32src(const float* __restrict__ x, const float* __restrict__ theta,
                      const float* __restrict__ W1, const float* __restrict__ W2,
                      float* __restrict__ out)
{
    __shared__ float    sW1[FFN_DIM * 4];
    __shared__ _Float16 sB[BN * BSTRIDE];

    const int tid  = threadIdx.x;
    const int lane = tid & 31;
    const int l16  = lane & 15;
    const int hk   = lane >> 4;
    const int wave = tid >> 5;

    const int m0 = blockIdx.x * BM;
    const int n0 = blockIdx.y * BN;

    for (int i = tid; i < FFN_DIM; i += 256)
        ((v4f*)sW1)[i] = ((const v4f*)W1)[i];

    float meas[4];
    {
        const int m = m0 + wave * 16 + l16;
        v4f xr = *(const v4f*)(x + (size_t)m * E_DIM);
        meas[0] = __cosf(xr[0]) * __cosf(theta[0]);
        meas[1] = __cosf(xr[1]) * __cosf(theta[1]);
        meas[2] = __cosf(xr[2]) * __cosf(theta[2]);
        meas[3] = __cosf(xr[3]) * __cosf(theta[3]);
    }

    v8f acc[8];
#pragma unroll
    for (int c = 0; c < 8; ++c)
#pragma unroll
        for (int r = 0; r < 8; ++r) acc[c][r] = 0.0f;

    const float* w2src = W2 + (size_t)(n0 + (tid >> 1)) * FFN_DIM + (tid & 1) * 16;
    _Float16*    bdst  = &sB[(tid >> 1) * BSTRIDE + (tid & 1) * 16];

    for (int kc = 0; kc < FFN_DIM; kc += BK) {
        __syncthreads();
        {
            const float* src = w2src + kc;
            v8h h0, h1;
#pragma unroll
            for (int j = 0; j < 8; ++j) {
                h0[j] = (_Float16)src[j];
                h1[j] = (_Float16)src[8 + j];
            }
            *(v8h*)bdst       = h0;
            *(v8h*)(bdst + 8) = h1;
            __builtin_prefetch(src + BK, 0, 0);
        }
        __syncthreads();
        mma_step(sW1, sB, meas, kc, hk, l16, acc);
    }

    const int mb = m0 + wave * 16 + 8 * hk;
#pragma unroll
    for (int c = 0; c < 8; ++c) {
        const int n = n0 + c * 16 + l16;
#pragma unroll
        for (int r = 0; r < 8; ++r)
            __builtin_nontemporal_store(acc[c][r],
                &out[(size_t)(mb + r) * E_DIM + n]);
    }
}

// ---- one-shot W2 f32 -> f16 conversion into workspace (2 MB) ----
__global__ __launch_bounds__(256)
void convert_w2_f16(const float* __restrict__ W2, _Float16* __restrict__ W2h)
{
    const int i = (blockIdx.x * 256 + threadIdx.x) * 8;
    v4f a = *(const v4f*)(W2 + i);
    v4f b = *(const v4f*)(W2 + i + 4);
    v8h o;
#pragma unroll
    for (int j = 0; j < 4; ++j) {
        o[j]     = (_Float16)a[j];
        o[4 + j] = (_Float16)b[j];
    }
    *(v8h*)(W2h + i) = o;
}

extern "C" void kernel_launch(void* const* d_in, const int* in_sizes, int n_in,
                              void* d_out, int out_size, void* d_ws, size_t ws_size,
                              hipStream_t stream) {
    const float* x     = (const float*)d_in[0];
    const float* theta = (const float*)d_in[1];
    const float* W1    = (const float*)d_in[2];
    const float* W2    = (const float*)d_in[3];
    float* out = (float*)d_out;

    dim3 grid(TOKENS / BM, E_DIM / BN);   // (256, 4)
    const size_t w2hBytes = (size_t)E_DIM * FFN_DIM * sizeof(_Float16);  // 2 MB

    if (ws_size >= w2hBytes) {
        _Float16* W2h = (_Float16*)d_ws;
        convert_w2_f16<<<dim3((E_DIM * FFN_DIM) / (256 * 8)), dim3(256), 0, stream>>>(W2, W2h);
        qffn_wmma_tdm<<<grid, dim3(256), 0, stream>>>(x, theta, W1, W2h, out);
    } else {
        qffn_wmma_f32src<<<grid, dim3(256), 0, stream>>>(x, theta, W1, W2, out);
    }

    (void)in_sizes; (void)n_in; (void)out_size;
}
